// ObTrEncoder_65678639891295
// MI455X (gfx1250) — matrix operand order repaired and checked by reading, not compiled, not run on hardware
//
#include <hip/hip_runtime.h>
#include <math.h>

#define DD 5
#define HH 24
#define WW 48
#define CL 1024
#define NB 4
#define DIM 128
#define DH 32
#define LMAX 12
#define DEPTH 6
#define LTOK 16            // NB + LMAX
#define BALL (DD*HH*WW)    // 5760
#define NHEAD (DIM/DH)     // 4

typedef __attribute__((ext_vector_type(16))) __bf16 v16bf;
typedef __attribute__((ext_vector_type(8)))  __bf16 v8bf;
typedef __attribute__((ext_vector_type(8)))  float  v8f;

// ---- WMMA fragment loaders per CDNA5 ISA 7.12.2 (wave32) ----
// 16-bit A 16x32, contiguous k order (not the ISA table order; B matches): lanes 0-15 -> M=lane, halves 0..15 = K 0..15
//                 lanes 16-31 -> M=lane-16, K 16..31
// -> two contiguous 16-byte LDS reads per lane.
__device__ __forceinline__ v16bf ldA(const __bf16* A, int lda, int kb, int lane) {
  int m = lane & 15; int khi = (lane >> 4) << 4;
  const __bf16* p = A + m * lda + kb + khi;
  v8bf lo = *(const v8bf*)p;
  v8bf hi = *(const v8bf*)(p + 8);
  return __builtin_shufflevector(lo, hi, 0, 1, 2, 3, 4, 5, 6, 7,
                                 8, 9, 10, 11, 12, 13, 14, 15);
}
// Pre-packed B fragments: dst[(tile*KS + ks)*512 + lane*16 + i] holds
// B[k][n] with n = nt*16 + (lane&15), k = ks*32 + ((lane>>4)<<4) + i.
// -> one contiguous 32-byte global read per lane (2x global_load_b128).
__device__ __forceinline__ v16bf ldBpk(const __bf16* Bp, int tile, int ks, int KS,
                                       int lane) {
  return *(const v16bf*)(Bp + ((((size_t)tile * KS) + ks) << 9) + lane * 16);
}
// B from a transposed source: B[k][n] = T[n][k]  (for q @ k^T), contiguous 32B.
__device__ __forceinline__ v16bf ldBt(const __bf16* T, int ldt, int lane) {
  int n = lane & 15; int k0 = (lane >> 4) << 4;
  return *(const v16bf*)(T + n * ldt + k0);
}
__device__ __forceinline__ v8f wmma_bf16(v16bf a, v16bf b, v8f c) {
  return __builtin_amdgcn_wmma_f32_16x16x32_bf16(false, a, false, b, (short)0, c,
                                                 false, false);
}
__device__ __forceinline__ float gelu_tanh(float x) {
  float x3 = x * x * x;
  return 0.5f * x * (1.0f + tanhf(0.7978845608028654f * (x + 0.044715f * x3)));
}

// ---------------- prep kernels ----------------
// Repack W (batch Lb of K x N row-major f32) into bf16 wave32 B-fragment tiles.
__global__ void k_pack_b(const float* __restrict__ src, __bf16* __restrict__ dst,
                         int Lb, int K, int N) {
  long total = (long)Lb * K * N;
  long idx = (long)blockIdx.x * blockDim.x + threadIdx.x;
  if (idx >= total) return;
  int i = (int)(idx & 15);
  int lane = (int)((idx >> 4) & 31);
  long tile = idx >> 9; // (l*NT + nt)*KS + ks
  int KS = K >> 5, NT = N >> 4;
  int ks = (int)(tile % KS);
  long t2 = tile / KS;
  int nt = (int)(t2 % NT);
  int l = (int)(t2 / NT);
  int k = ks * 32 + ((lane >> 4) << 4) + i;
  int n = nt * 16 + (lane & 15);
  dst[idx] = (__bf16)src[((long)l * K + k) * N + n];
}
__global__ void k_zero_i32(int* p, int n) {
  int i = blockIdx.x * blockDim.x + threadIdx.x;
  if (i < n) p[i] = 0;
}
// constant "direct" vector: mean over NB of (ob_background @ direct_w) + direct_b
__global__ void k_constvec(const float* __restrict__ obbg, const float* __restrict__ dw,
                           const float* __restrict__ db, float* __restrict__ cv) {
  int j = blockIdx.x * blockDim.x + threadIdx.x;
  if (j >= NB * DIM) return;
  float s = db[j];
  for (int k = 0; k < DIM; ++k) {
    float ob = 0.25f * (obbg[k] + obbg[DIM + k] + obbg[2 * DIM + k] + obbg[3 * DIM + k]);
    s += ob * dw[k * (NB * DIM) + j];
  }
  cv[j] = s;
}
__global__ void k_init_tr(float* __restrict__ tr, const float* __restrict__ obbg,
                          const float* __restrict__ cv) {
  long idx = (long)blockIdx.x * blockDim.x + threadIdx.x;
  if (idx >= (long)BALL * LTOK * DIM) return;
  int d = (int)(idx % DIM);
  int slot = (int)((idx / DIM) % LTOK);
  float v = 0.f;
  if (slot < NB) v = obbg[slot * DIM + d] + cv[slot * DIM + d];
  tr[idx] = v;
}
__global__ void k_scatter(const float* __restrict__ x, const int* __restrict__ li,
                          float* __restrict__ tr, int* __restrict__ counts, int N) {
  int i = blockIdx.x * blockDim.x + threadIdx.x;
  if (i >= N) return;
  int d = li[3 * i], h = li[3 * i + 1], w = li[3 * i + 2];
  int flat = (d * HH + h) * WW + w;
  int p = atomicAdd(&counts[flat], 1);
  if (p < LMAX) {
    float* dst = tr + ((size_t)flat * LTOK + NB + p) * DIM;
    const float* src = x + (size_t)i * DIM;
    for (int k = 0; k < DIM; ++k) dst[k] = src[k];
  }
}

// ---------------- main transformer: one workgroup (4 wave32) per ball ----------------
__global__ __launch_bounds__(128)
void k_transformer(const float* __restrict__ tr,
                   const float* __restrict__ ln1_w, const float* __restrict__ ln1_b,
                   const __bf16* __restrict__ wqkvp, const float* __restrict__ bqkv,
                   const __bf16* __restrict__ wop, const float* __restrict__ bo,
                   const float* __restrict__ ln2_w, const float* __restrict__ ln2_b,
                   const __bf16* __restrict__ w1p, const float* __restrict__ b1,
                   const __bf16* __restrict__ w2p, const float* __restrict__ b2,
                   __bf16* __restrict__ hnb) {
  __shared__ float  hs[LTOK][DIM];          // fp32 residual stream
  __shared__ __bf16 xb[LTOK][DIM];          // LN output / attn output (A operand)
  __shared__ __bf16 qkvb[LTOK][3 * DIM];    // q|k|v bf16
  __shared__ float  att[NHEAD][LTOK][LTOK]; // scores / probs
  __shared__ __bf16 ub[LTOK][4 * DIM];      // MLP hidden
  __shared__ float  mu[LTOK], rs[LTOK];

  const int ball = blockIdx.x;
  const int tid = threadIdx.x;
  const int lane = tid & 31;
  const int wave = tid >> 5;
  const float scale = 0.17677669529663687f; // 1/sqrt(32)

  for (int i = tid; i < LTOK * DIM; i += 128)
    hs[i >> 7][i & 127] = tr[(size_t)ball * (LTOK * DIM) + i];
  __syncthreads();

  for (int layer = 0; layer < DEPTH; ++layer) {
    const __bf16* Wq = wqkvp + (size_t)layer * (24 * 4 * 512);
    const __bf16* Wo = wop + (size_t)layer * (8 * 4 * 512);
    const __bf16* W1 = w1p + (size_t)layer * (32 * 4 * 512);
    const __bf16* W2 = w2p + (size_t)layer * (8 * 16 * 512);

    // ---- LN1 -> xb ----
    if (tid < LTOK) {
      float m = 0.f;
      for (int d = 0; d < DIM; ++d) m += hs[tid][d];
      m *= (1.f / DIM);
      float v = 0.f;
      for (int d = 0; d < DIM; ++d) { float t = hs[tid][d] - m; v += t * t; }
      mu[tid] = m; rs[tid] = rsqrtf(v * (1.f / DIM) + 1e-5f);
    }
    __syncthreads();
    for (int i = tid; i < LTOK * DIM; i += 128) {
      int l = i >> 7, d = i & 127;
      xb[l][d] = (__bf16)((hs[l][d] - mu[l]) * rs[l] * ln1_w[layer * DIM + d] +
                          ln1_b[layer * DIM + d]);
    }
    __syncthreads();

    // ---- QKV: (16x128)@(128x384), 24 N-tiles over 4 waves ----
    for (int t = wave; t < 24; t += 4) {
      if (t + 4 < 24)
        __builtin_prefetch(Wq + (((size_t)(t + 4) * 4) << 9) + lane * 16, 0, 3);
      v8f c = {};
      #pragma unroll
      for (int ks = 0; ks < 4; ++ks) {
        v16bf a = ldA(&xb[0][0], DIM, ks * 32, lane);
        v16bf b = ldBpk(Wq, t, ks, 4, lane);
        c = wmma_bf16(a, b, c);
      }
      int n = t * 16 + (lane & 15), mb = (lane >> 4) << 3;
      #pragma unroll
      for (int r = 0; r < 8; ++r)
        qkvb[mb + r][n] = (__bf16)(c[r] + bqkv[layer * 3 * DIM + n]);
    }
    __syncthreads();

    // ---- scores: one head per wave, single K=32 WMMA: q_h @ k_h^T ----
    {
      int hh = wave;
      v16bf a = ldA(&qkvb[0][0] + hh * DH, 3 * DIM, 0, lane);
      v16bf b = ldBt(&qkvb[0][0] + DIM + hh * DH, 3 * DIM, lane);
      v8f c = {};
      c = wmma_bf16(a, b, c);
      int n = lane & 15, mb = (lane >> 4) << 3;
      #pragma unroll
      for (int r = 0; r < 8; ++r) att[hh][mb + r][n] = c[r] * scale;
    }
    __syncthreads();

    // ---- softmax over 16 keys, one thread per (head, query) ----
    if (tid < NHEAD * LTOK) {
      int hh = tid >> 4, l = tid & 15;
      float mx = -1e30f;
      for (int m2 = 0; m2 < LTOK; ++m2) mx = fmaxf(mx, att[hh][l][m2]);
      float s = 0.f;
      for (int m2 = 0; m2 < LTOK; ++m2) {
        float e = __expf(att[hh][l][m2] - mx);
        att[hh][l][m2] = e; s += e;
      }
      float inv = 1.f / s;
      for (int m2 = 0; m2 < LTOK; ++m2) att[hh][l][m2] *= inv;
    }
    __syncthreads();

    // ---- o = att @ v (tiny: 16 fma per element), write to xb as next A ----
    for (int i = tid; i < LTOK * DIM; i += 128) {
      int l = i >> 7, dd = i & 127, hh = dd >> 5, d = dd & 31;
      float s = 0.f;
      for (int m2 = 0; m2 < LTOK; ++m2)
        s += att[hh][l][m2] * (float)qkvb[m2][2 * DIM + hh * DH + d];
      xb[l][dd] = (__bf16)s;
    }
    __syncthreads();

    // ---- out-proj + residual: (16x128)@(128x128) ----
    for (int t = wave; t < 8; t += 4) {
      v8f c = {};
      #pragma unroll
      for (int ks = 0; ks < 4; ++ks) {
        v16bf a = ldA(&xb[0][0], DIM, ks * 32, lane);
        v16bf b = ldBpk(Wo, t, ks, 4, lane);
        c = wmma_bf16(a, b, c);
      }
      int n = t * 16 + (lane & 15), mb = (lane >> 4) << 3;
      #pragma unroll
      for (int r = 0; r < 8; ++r) hs[mb + r][n] += c[r] + bo[layer * DIM + n];
    }
    __syncthreads();

    // ---- LN2 -> xb ----
    if (tid < LTOK) {
      float m = 0.f;
      for (int d = 0; d < DIM; ++d) m += hs[tid][d];
      m *= (1.f / DIM);
      float v = 0.f;
      for (int d = 0; d < DIM; ++d) { float t = hs[tid][d] - m; v += t * t; }
      mu[tid] = m; rs[tid] = rsqrtf(v * (1.f / DIM) + 1e-5f);
    }
    __syncthreads();
    for (int i = tid; i < LTOK * DIM; i += 128) {
      int l = i >> 7, d = i & 127;
      xb[l][d] = (__bf16)((hs[l][d] - mu[l]) * rs[l] * ln2_w[layer * DIM + d] +
                          ln2_b[layer * DIM + d]);
    }
    __syncthreads();

    // ---- MLP1 + gelu: (16x128)@(128x512) ----
    for (int t = wave; t < 32; t += 4) {
      if (t + 4 < 32)
        __builtin_prefetch(W1 + (((size_t)(t + 4) * 4) << 9) + lane * 16, 0, 3);
      v8f c = {};
      #pragma unroll
      for (int ks = 0; ks < 4; ++ks) {
        v16bf a = ldA(&xb[0][0], DIM, ks * 32, lane);
        v16bf b = ldBpk(W1, t, ks, 4, lane);
        c = wmma_bf16(a, b, c);
      }
      int n = t * 16 + (lane & 15), mb = (lane >> 4) << 3;
      #pragma unroll
      for (int r = 0; r < 8; ++r)
        ub[mb + r][n] = (__bf16)gelu_tanh(c[r] + b1[layer * 4 * DIM + n]);
    }
    __syncthreads();

    // ---- MLP2 + residual: (16x512)@(512x128) ----
    for (int t = wave; t < 8; t += 4) {
      v8f c = {};
      #pragma unroll
      for (int ks = 0; ks < 16; ++ks) {
        v16bf a = ldA(&ub[0][0], 4 * DIM, ks * 32, lane);
        v16bf b = ldBpk(W2, t, ks, 16, lane);
        c = wmma_bf16(a, b, c);
      }
      int n = t * 16 + (lane & 15), mb = (lane >> 4) << 3;
      #pragma unroll
      for (int r = 0; r < 8; ++r) hs[mb + r][n] += c[r] + b2[layer * DIM + n];
    }
    __syncthreads();
  }

  // export first NB tokens as bf16 for the final GEMM
  for (int i = tid; i < NB * DIM; i += 128)
    hnb[(size_t)ball * (NB * DIM) + i] = (__bf16)hs[i >> 7][i & 127];
}

// ---------------- final projection: (5760x512)@(512x1024) ----------------
__global__ __launch_bounds__(128)
void k_final(const __bf16* __restrict__ hnb, const __bf16* __restrict__ Wp,
             const float* __restrict__ bias, const int* __restrict__ counts,
             float* __restrict__ out) {
  __shared__ __bf16 As[16][NB * DIM]; // 16 balls x 512
  int tid = threadIdx.x, lane = tid & 31, wave = tid >> 5;
  int rowbase = blockIdx.x * 16;
  for (int i = tid; i < 16 * NB * DIM; i += 128)
    As[i >> 9][i & 511] = hnb[(size_t)(rowbase + (i >> 9)) * (NB * DIM) + (i & 511)];
  __syncthreads();
  int nb0 = blockIdx.y * 256 + wave * 64;
  for (int j = 0; j < 4; ++j) {
    int nt = (nb0 >> 4) + j;
    if (j + 1 < 4)
      __builtin_prefetch(Wp + (((size_t)(nt + 1) * 16) << 9) + lane * 16, 0, 3);
    v8f c = {};
    #pragma unroll
    for (int ks = 0; ks < 16; ++ks) {
      v16bf a = ldA(&As[0][0], NB * DIM, ks * 32, lane);
      v16bf b = ldBpk(Wp, nt, ks, 16, lane);
      c = wmma_bf16(a, b, c);
    }
    int n = nt * 16 + (lane & 15), mb = (lane >> 4) << 3;
    #pragma unroll
    for (int r = 0; r < 8; ++r) {
      int ball = rowbase + mb + r;
      float v = c[r] + bias[n];
      out[(size_t)ball * CL + n] = (counts[ball] > 0) ? v : 0.f;
    }
  }
}

extern "C" void kernel_launch(void* const* d_in, const int* in_sizes, int n_in,
                              void* d_out, int out_size, void* d_ws, size_t ws_size,
                              hipStream_t stream) {
  (void)n_in; (void)out_size; (void)ws_size;
  const float* x        = (const float*)d_in[0];
  const int*   li       = (const int*)d_in[1];
  const float* obbg     = (const float*)d_in[2];
  const float* direct_w = (const float*)d_in[3];
  const float* direct_b = (const float*)d_in[4];
  const float* ln1_w    = (const float*)d_in[5];
  const float* ln1_b    = (const float*)d_in[6];
  const float* wqkv     = (const float*)d_in[7];
  const float* bqkv     = (const float*)d_in[8];
  const float* wo       = (const float*)d_in[9];
  const float* bo       = (const float*)d_in[10];
  const float* ln2_w    = (const float*)d_in[11];
  const float* ln2_b    = (const float*)d_in[12];
  const float* w1       = (const float*)d_in[13];
  const float* b1       = (const float*)d_in[14];
  const float* w2       = (const float*)d_in[15];
  const float* b2       = (const float*)d_in[16];
  const float* ob2l_w   = (const float*)d_in[17];
  const float* ob2l_b   = (const float*)d_in[18];
  const int N = in_sizes[0] / DIM;

  char* ws = (char*)d_ws;
  size_t off = 0;
  auto alloc = [&](size_t bytes) {
    size_t o = off; off += (bytes + 255) & ~(size_t)255; return o;
  };
  float*  tr     = (float*)(ws + alloc((size_t)BALL * LTOK * DIM * 4));
  int*    counts = (int*)(ws + alloc((size_t)BALL * 4));
  float*  cv     = (float*)(ws + alloc((size_t)NB * DIM * 4));
  __bf16* wqkvp  = (__bf16*)(ws + alloc((size_t)DEPTH * DIM * 3 * DIM * 2));
  __bf16* wop    = (__bf16*)(ws + alloc((size_t)DEPTH * DIM * DIM * 2));
  __bf16* w1p    = (__bf16*)(ws + alloc((size_t)DEPTH * DIM * 4 * DIM * 2));
  __bf16* w2p    = (__bf16*)(ws + alloc((size_t)DEPTH * 4 * DIM * DIM * 2));
  __bf16* ob2lp  = (__bf16*)(ws + alloc((size_t)NB * DIM * CL * 2));
  __bf16* hnb    = (__bf16*)(ws + alloc((size_t)BALL * NB * DIM * 2));

  auto pack = [&](const float* s, __bf16* d, int Lb, int K, int Ncols) {
    long n = (long)Lb * K * Ncols;
    k_pack_b<<<(int)((n + 255) / 256), 256, 0, stream>>>(s, d, Lb, K, Ncols);
  };
  pack(wqkv,   wqkvp, DEPTH, DIM, 3 * DIM);
  pack(wo,     wop,   DEPTH, DIM, DIM);
  pack(w1,     w1p,   DEPTH, DIM, 4 * DIM);
  pack(w2,     w2p,   DEPTH, 4 * DIM, DIM);
  pack(ob2l_w, ob2lp, 1, NB * DIM, CL);

  k_constvec<<<2, 256, 0, stream>>>(obbg, direct_w, direct_b, cv);
  long trtot = (long)BALL * LTOK * DIM;
  k_init_tr<<<(int)((trtot + 255) / 256), 256, 0, stream>>>(tr, obbg, cv);
  k_zero_i32<<<(BALL + 255) / 256, 256, 0, stream>>>(counts, BALL);
  k_scatter<<<(N + 255) / 256, 256, 0, stream>>>(x, li, tr, counts, N);
  k_transformer<<<BALL, 128, 0, stream>>>(tr, ln1_w, ln1_b, wqkvp, bqkv, wop, bo,
                                          ln2_w, ln2_b, w1p, b1, w2p, b2, hnb);
  k_final<<<dim3(BALL / 16, CL / 256), 128, 0, stream>>>(hnb, ob2lp, ob2l_b, counts,
                                                         (float*)d_out);
}
